// Encoder_89507118448901
// MI455X (gfx1250) — matrix operand order repaired and checked by reading, not compiled, not run on hardware
//
#include <hip/hip_runtime.h>
#include <hip/hip_bf16.h>

typedef _Float16 half_t;
typedef __attribute__((ext_vector_type(16))) _Float16 v16h;
typedef __attribute__((ext_vector_type(8)))  _Float16 v8h;
typedef __attribute__((ext_vector_type(8)))  float    v8f;
typedef __attribute__((ext_vector_type(2)))  float    v2f;
typedef __attribute__((ext_vector_type(4)))  unsigned int u32x4;
typedef __attribute__((ext_vector_type(4)))  int      i32x4;
typedef __attribute__((ext_vector_type(8)))  int      i32x8;

#define Bsz   256
#define Tlen  512
#define Edim  200
#define Hdim  100
#define Gdim  400      // 4*H
#define Kpad  224      // K padded to 7*32 for f16 WMMA
#define Mrows (Bsz*Tlen)   // 131072
#define BB    32       // batch slice per recurrence block

// ---------------------------------------------------------------------------
// Weight prep: transpose (K,N)->(N,Kpad) f16 with zero pad; (K,N)->(N,K) f32
// ---------------------------------------------------------------------------
__global__ void transpose_pad_f16_kernel(half_t* __restrict__ dst,
                                         const float* __restrict__ src,
                                         int K, int N, int KpadA) {
  int tid = blockIdx.x * blockDim.x + threadIdx.x;
  if (tid >= N * KpadA) return;
  int n = tid / KpadA, k = tid % KpadA;
  dst[tid] = (k < K) ? (half_t)src[(size_t)k * N + n] : (half_t)0.0f;
}

__global__ void transpose_f32_kernel(float* __restrict__ dst,
                                     const float* __restrict__ src,
                                     int K, int N) {
  int tid = blockIdx.x * blockDim.x + threadIdx.x;
  if (tid >= N * K) return;
  int n = tid / K, k = tid % K;
  dst[tid] = src[(size_t)k * N + n];
}

// ---------------------------------------------------------------------------
// Embedding gather -> f16, rows [m][Kpad], zero pad beyond E
// ---------------------------------------------------------------------------
__global__ void embed_kernel(const int* __restrict__ x,
                             const float* __restrict__ emb,
                             half_t* __restrict__ e16) {
  long long tid = (long long)blockIdx.x * blockDim.x + threadIdx.x;
  if (tid >= (long long)Mrows * (Kpad / 8)) return;
  int m  = (int)(tid / (Kpad / 8));
  int k8 = (int)(tid % (Kpad / 8)) * 8;
  int token = x[m];
  v8h out;
#pragma unroll
  for (int j = 0; j < 8; ++j) {
    int k = k8 + j;
    out[j] = (k < Edim) ? (half_t)emb[(size_t)token * Edim + k] : (half_t)0.0f;
  }
  *(v8h*)(e16 + (size_t)m * Kpad + k8) = out;
}

// zero the K-pad columns [200,224) of seq1_f16
__global__ void padzero_kernel(half_t* __restrict__ seq16) {
  int tid = blockIdx.x * blockDim.x + threadIdx.x;
  if (tid >= Mrows * 3) return;
  int m  = tid / 3;
  int k8 = 200 + (tid % 3) * 8;
  v8h z = {(half_t)0.f,(half_t)0.f,(half_t)0.f,(half_t)0.f,
           (half_t)0.f,(half_t)0.f,(half_t)0.f,(half_t)0.f};
  *(v8h*)(seq16 + (size_t)m * Kpad + k8) = z;
}

// ---------------------------------------------------------------------------
// xz GEMM:  out[t][b][400] = A[m][Kpad] @ WxT[n][Kpad] + bias[n]   (f16 WMMA)
// block: 256 threads = 8 waves; block tile M=128, N=80; wave tile 16x80.
// B slab (80x224 f16, 35.8KB) staged once into LDS (shared by all 8 waves).
// Per k-chunk: load all 5 B fragments first, then 5 independent WMMAs, so
// the scheduler can use partial s_wait_dscnt instead of full serialization.
// blockIdx.z selects direction (fwd/bwd weights share A).
// ---------------------------------------------------------------------------
__global__ __launch_bounds__(256) void gemm_xz_kernel(
    const half_t* __restrict__ A,
    const half_t* __restrict__ WT0, const half_t* __restrict__ WT1,
    const float*  __restrict__ bias0, const float* __restrict__ bias1,
    half_t* __restrict__ out0, half_t* __restrict__ out1) {
  const half_t* WT  = blockIdx.z ? WT1 : WT0;
  const float*  bia = blockIdx.z ? bias1 : bias0;
  half_t*       out = blockIdx.z ? out1 : out0;

  __shared__ half_t sB[80 * Kpad];            // 35840 B

  int lane  = threadIdx.x & 31;
  int wave  = threadIdx.x >> 5;
  int n16   = lane & 15;
  int khalf = lane >> 4;
  int rowA  = blockIdx.x * 128 + wave * 16 + n16;   // A row this lane reads
  int bn    = blockIdx.y * 80;

  // stage B slab into LDS (rows bn..bn+79 are contiguous in WT)
  for (int j = threadIdx.x; j < 80 * Kpad / 8; j += 256)
    *(v8h*)(sB + j * 8) = *(const v8h*)(WT + (size_t)bn * Kpad + j * 8);
  __syncthreads();

  v8f acc[5];
  float bv[5];
#pragma unroll
  for (int j = 0; j < 5; ++j) {
    acc[j] = (v8f){0.f,0.f,0.f,0.f,0.f,0.f,0.f,0.f};
    bv[j]  = bia[bn + j * 16 + n16];
  }

  for (int kc = 0; kc < 7; ++kc) {
    union { v16h v; v8h h[2]; } af;
    const half_t* ap = A + (size_t)rowA * Kpad + kc * 32 + khalf * 16;
    af.h[0] = *(const v8h*)(ap);
    af.h[1] = *(const v8h*)(ap + 8);
    if (kc < 6) __builtin_prefetch(ap + 32, 0, 3);   // global_prefetch_b8 (near)

    v16h bf[5];
#pragma unroll
    for (int j = 0; j < 5; ++j)
      bf[j] = *(const v16h*)(sB + (size_t)(j * 16 + n16) * Kpad + kc * 32 + khalf * 16);
#pragma unroll
    for (int j = 0; j < 5; ++j)
      acc[j] = __builtin_amdgcn_wmma_f32_16x16x32_f16(
          false, af.v, false, bf[j], (short)0, acc[j], false, false);
  }

#pragma unroll
  for (int j = 0; j < 5; ++j) {
#pragma unroll
    for (int v = 0; v < 8; ++v) {
      int m = blockIdx.x * 128 + wave * 16 + v + 8 * khalf;   // C: M = v + 8*khalf
      int b = m >> 9, t = m & 511;
      size_t idx = ((size_t)t * Bsz + b) * Gdim + bn + j * 16 + n16;
      out[idx] = (half_t)(acc[j][v] + bv[j]);
    }
  }
}

// ---------------------------------------------------------------------------
// LSTM recurrence. One block = one direction x 32-batch slice, loops all T.
// LDS: Wh^T (400x100 f32, TDM-loaded), z (32x400), h/c (32x104), mask (16KB).
// f32 WMMA 16x16x4, K=100 split into 2 chains x 2 paired tiles (4-way ILP).
// grid.x = 2*(B/BB): even = forward, odd = backward
// ---------------------------------------------------------------------------
#define REC_LDS_FLOATS (40000 + 12800 + 3328 + 3328)
#define REC_LDS_BYTES  (REC_LDS_FLOATS * 4 + BB * Tlen)    // + mask bytes

__global__ __launch_bounds__(256) void lstm_rec_kernel(
    const half_t* __restrict__ xz_f, const half_t* __restrict__ xz_b,
    const float*  __restrict__ WhT_f, const float* __restrict__ WhT_b,
    const int*    __restrict__ xtok,
    half_t* __restrict__ seq_out,            // may be null
    float* __restrict__ hf_out, float* __restrict__ hb_out,
    int write_seq) {
  extern __shared__ float smem[];
  float* sWh = smem;                 // 400*100 = 40000 floats (at LDS offset 0)
  float* sZ  = smem + 40000;         // 32*400 = 12800
  float* sH  = smem + 52800;         // 32*104 = 3328
  float* sC  = smem + 56128;         // 32*104 = 3328
  char*  sM  = (char*)(smem + REC_LDS_FLOATS);   // 32*512 mask bytes

  int tid   = threadIdx.x;
  int dir   = blockIdx.x & 1;
  int bbase = (blockIdx.x >> 1) * BB;
  const half_t* xz  = dir ? xz_b  : xz_f;
  const float*  WhT = dir ? WhT_b : WhT_f;
  float*        hou = dir ? hb_out : hf_out;
  int coloff = dir ? Hdim : 0;

  // ---- stage Wh^T into LDS: TDM if available, else cooperative copy ----
#if __has_builtin(__builtin_amdgcn_tensor_load_to_lds) && __has_builtin(__builtin_amdgcn_s_wait_tensorcnt)
  if (tid < 32) {
    unsigned long long ga = (unsigned long long)WhT;
    u32x4 g0 = { 1u,                                   // count=1, user mode
                 0u,                                   // lds_addr = 0 (sWh at base)
                 (unsigned int)ga,                     // global_addr[31:0]
                 (unsigned int)(((ga >> 32) & 0x1FFFFFFu) | (2u << 30)) }; // addr[56:32] | type=2
    i32x8 g1 = { (int)(2u << 16),          // data_size=4B (code 2), mask=0
                 (int)(40000u << 16),      // tensor_dim0[15:0] = 40000
                 (int)(1u << 16),          // dim0[31:16]=0 | tensor_dim1=1
                 (int)(40000u << 16),      // dim1 hi=0 | tile_dim0 = 40000
                 1,                        // tile_dim1=1, tile_dim2=0
                 40000,                    // tensor_dim0_stride[31:0]
                 0, 0 };
    i32x4 z4 = {0, 0, 0, 0};
#if __has_include(<hip/amd_detail/amd_gfx1250_TDM.h>)
    i32x8 z8 = {0, 0, 0, 0, 0, 0, 0, 0};
    __builtin_amdgcn_tensor_load_to_lds(g0, g1, z4, z4, z8, 0);   // 6-arg toolchain
#else
    __builtin_amdgcn_tensor_load_to_lds(g0, g1, z4, z4, 0);       // 5-arg toolchain
#endif
    __builtin_amdgcn_s_wait_tensorcnt(0);
  }
#else
  for (int i = tid; i < Gdim * Hdim; i += 256) sWh[i] = WhT[i];
#endif

  // h/c init + mask stage (removes per-step global token loads)
  for (int i = tid; i < BB * 104; i += 256) { sH[i] = 0.f; sC[i] = 0.f; }
  for (int i = tid; i < BB * Tlen; i += 256) {
    int bl = i >> 9, tt = i & 511;
    sM[i] = (char)(xtok[(bbase + bl) * Tlen + tt] != 0);
  }
  __syncthreads();

  int lane = tid & 31, wave = tid >> 5;
  int n16 = lane & 15, khalf = lane >> 4;
  const v8f zero8 = (v8f){0.f,0.f,0.f,0.f,0.f,0.f,0.f,0.f};

  for (int s = 0; s < Tlen; ++s) {
    int t = dir ? (Tlen - 1 - s) : s;

    // ---- z = xz[t] + h @ Wh : per tn, paired tm tiles + split-K chains ----
    for (int tn = wave; tn < 25; tn += 8) {
      int bn = tn * 16 + n16;                   // Wh^T row for this lane
      // xz preload (consumed only at the store -> overlaps WMMA chain)
      float xv0[8], xv1[8];
      const half_t* xzp = xz + ((size_t)t * Bsz + bbase) * Gdim + tn * 16 + n16;
#pragma unroll
      for (int v = 0; v < 8; ++v) {
        xv0[v] = (float)xzp[(size_t)(v + 8 * khalf) * Gdim];
        xv1[v] = (float)xzp[(size_t)(16 + v + 8 * khalf) * Gdim];
      }
      v8f acc00 = zero8, acc01 = zero8, acc10 = zero8, acc11 = zero8;
#pragma unroll
      for (int kk = 0; kk < 13; ++kk) {
        v2f bf = *(const v2f*)&sWh[bn * Hdim + kk * 4 + 2 * khalf];
        v2f a0 = *(const v2f*)&sH [n16 * 104        + kk * 4 + 2 * khalf];
        v2f a1 = *(const v2f*)&sH [(16 + n16) * 104 + kk * 4 + 2 * khalf];
        acc00 = __builtin_amdgcn_wmma_f32_16x16x4_f32(false, a0, false, bf, (short)0, acc00, false, false);
        acc10 = __builtin_amdgcn_wmma_f32_16x16x4_f32(false, a1, false, bf, (short)0, acc10, false, false);
        if (kk < 12) {
          int k2 = kk + 13;
          v2f bg = *(const v2f*)&sWh[bn * Hdim + k2 * 4 + 2 * khalf];
          v2f a2 = *(const v2f*)&sH [n16 * 104        + k2 * 4 + 2 * khalf];
          v2f a3 = *(const v2f*)&sH [(16 + n16) * 104 + k2 * 4 + 2 * khalf];
          acc01 = __builtin_amdgcn_wmma_f32_16x16x4_f32(false, a2, false, bg, (short)0, acc01, false, false);
          acc11 = __builtin_amdgcn_wmma_f32_16x16x4_f32(false, a3, false, bg, (short)0, acc11, false, false);
        }
      }
#pragma unroll
      for (int v = 0; v < 8; ++v) {
        sZ[(v + 8 * khalf) * Gdim      + tn * 16 + n16] = acc00[v] + acc01[v] + xv0[v];
        sZ[(16 + v + 8 * khalf) * Gdim + tn * 16 + n16] = acc10[v] + acc11[v] + xv1[v];
      }
    }
    __syncthreads();

    // ---- gates + masked state update (mask from LDS) ----
    for (int e = tid; e < BB * Hdim; e += 256) {
      int bl = e / Hdim, k = e % Hdim;
      float zi = sZ[bl * Gdim + k];
      float zf = sZ[bl * Gdim + Hdim + k];
      float zg = sZ[bl * Gdim + 2 * Hdim + k];
      float zo = sZ[bl * Gdim + 3 * Hdim + k];
      float ig = 1.f / (1.f + __expf(-zi));
      float fg = 1.f / (1.f + __expf(-zf));
      float gg = 1.f - 2.f / (__expf(2.f * zg) + 1.f);   // tanh
      float og = 1.f / (1.f + __expf(-zo));
      float cold = sC[bl * 104 + k];
      float cnew = fg * cold + ig * gg;
      float hnew = og * (1.f - 2.f / (__expf(2.f * cnew) + 1.f));
      bool  msk  = sM[bl * Tlen + t] != 0;
      float hold = sH[bl * 104 + k];
      float hh = msk ? hnew : hold;
      float cc = msk ? cnew : cold;
      sH[bl * 104 + k] = hh;
      sC[bl * 104 + k] = cc;
      if (write_seq)
        seq_out[((size_t)(bbase + bl) * Tlen + t) * Kpad + coloff + k] = (half_t)hh;
    }
    __syncthreads();
  }

  for (int e = tid; e < BB * Hdim; e += 256) {
    int bl = e / Hdim, k = e % Hdim;
    hou[(bbase + bl) * Hdim + k] = sH[bl * 104 + k];
  }
}

// ---------------------------------------------------------------------------
// Dense heads: out[head][b][600] = concat(hf,hb) @ dW + db
// ---------------------------------------------------------------------------
__global__ __launch_bounds__(640) void head_kernel(
    const float* __restrict__ hf1, const float* __restrict__ hb1,
    const float* __restrict__ hf2, const float* __restrict__ hb2,
    const float* __restrict__ d1W, const float* __restrict__ d1b,
    const float* __restrict__ d2W, const float* __restrict__ d2b,
    float* __restrict__ out) {
  __shared__ float hs[2 * Hdim];
  int b = blockIdx.x, head = blockIdx.y, tid = threadIdx.x;
  const float* hf = head ? hf2 : hf1;
  const float* hb = head ? hb2 : hb1;
  const float* W  = head ? d2W : d1W;
  const float* bi = head ? d2b : d1b;
  if (tid < 200) hs[tid] = (tid < Hdim) ? hf[b * Hdim + tid]
                                        : hb[b * Hdim + tid - Hdim];
  __syncthreads();
  if (tid < 600) {
    float sum = bi[tid];
    for (int k = 0; k < 2 * Hdim; ++k) sum += hs[k] * W[(size_t)k * 600 + tid];
    out[(size_t)head * (Bsz * 600) + (size_t)b * 600 + tid] = sum;
  }
}

// ---------------------------------------------------------------------------
extern "C" void kernel_launch(void* const* d_in, const int* in_sizes, int n_in,
                              void* d_out, int out_size, void* d_ws, size_t ws_size,
                              hipStream_t stream) {
  const int*   x    = (const int*)d_in[0];
  const float* emb  = (const float*)d_in[1];
  const float* Wx[4] = {(const float*)d_in[2],  (const float*)d_in[5],
                        (const float*)d_in[8],  (const float*)d_in[11]};
  const float* Wh[4] = {(const float*)d_in[3],  (const float*)d_in[6],
                        (const float*)d_in[9],  (const float*)d_in[12]};
  const float* bb[4] = {(const float*)d_in[4],  (const float*)d_in[7],
                        (const float*)d_in[10], (const float*)d_in[13]};
  const float* d1W = (const float*)d_in[14];
  const float* d1b = (const float*)d_in[15];
  const float* d2W = (const float*)d_in[16];
  const float* d2b = (const float*)d_in[17];
  float* out = (float*)d_out;

  char* ws = (char*)d_ws;
  size_t off = 0;
  auto carve = [&](size_t bytes) -> void* {
    void* p = ws + off;
    off += (bytes + 255) & ~(size_t)255;
    return p;
  };
  half_t* e16   = (half_t*)carve((size_t)Mrows * Kpad * 2);
  half_t* seq16 = (half_t*)carve((size_t)Mrows * Kpad * 2);
  half_t* wxT[4]; for (int i = 0; i < 4; ++i) wxT[i] = (half_t*)carve((size_t)Gdim * Kpad * 2);
  float*  whT[4]; for (int i = 0; i < 4; ++i) whT[i] = (float*)carve((size_t)Gdim * Hdim * 4);
  half_t* xzb[4]; for (int i = 0; i < 4; ++i) xzb[i] = (half_t*)carve((size_t)Tlen * Bsz * Gdim * 2);
  float*  hl[4];  for (int i = 0; i < 4; ++i) hl[i]  = (float*)carve((size_t)Bsz * Hdim * 4);

  // ---- weight prep ----
  {
    int n = Gdim * Kpad, g = (n + 255) / 256;
    for (int i = 0; i < 4; ++i) {
      int K = (i < 2) ? Edim : 2 * Hdim;   // both 200
      transpose_pad_f16_kernel<<<g, 256, 0, stream>>>(wxT[i], Wx[i], K, Gdim, Kpad);
    }
    int n2 = Gdim * Hdim, g2 = (n2 + 255) / 256;
    for (int i = 0; i < 4; ++i)
      transpose_f32_kernel<<<g2, 256, 0, stream>>>(whT[i], Wh[i], Hdim, Gdim);
  }

  // ---- embedding + pad ----
  {
    long long n = (long long)Mrows * (Kpad / 8);
    embed_kernel<<<(int)((n + 255) / 256), 256, 0, stream>>>(x, emb, e16);
    int n2 = Mrows * 3;
    padzero_kernel<<<(n2 + 255) / 256, 256, 0, stream>>>(seq16);
  }

  // ---- layer 1 ----
  gemm_xz_kernel<<<dim3(Mrows / 128, Gdim / 80, 2), 256, 0, stream>>>(
      e16, wxT[0], wxT[1], bb[0], bb[1], xzb[0], xzb[1]);
  lstm_rec_kernel<<<2 * (Bsz / BB), 256, REC_LDS_BYTES, stream>>>(
      xzb[0], xzb[1], whT[0], whT[1], x, seq16, hl[0], hl[1], 1);

  // ---- layer 2 ----
  gemm_xz_kernel<<<dim3(Mrows / 128, Gdim / 80, 2), 256, 0, stream>>>(
      seq16, wxT[2], wxT[3], bb[2], bb[3], xzb[2], xzb[3]);
  lstm_rec_kernel<<<2 * (Bsz / BB), 256, REC_LDS_BYTES, stream>>>(
      xzb[2], xzb[3], whT[2], whT[3], x, (half_t*)nullptr, hl[2], hl[3], 0);

  // ---- heads ----
  head_kernel<<<dim3(Bsz, 2), 640, 0, stream>>>(
      hl[0], hl[1], hl[2], hl[3], d1W, d1b, d2W, d2b, out);
}
